// SetAbstraction_88235808129097
// MI455X (gfx1250) — matrix operand order repaired and checked by reading, not compiled, not run on hardware
//
#include <hip/hip_runtime.h>

#define Bv   4
#define Nv   8192
#define Cv   128
#define Sv   2048
#define Kv   32
#define CIN  131
#define CINP 160          // 131 padded to 5 * 32
#define DOUT 256
#define R2v  0.01f        // RADIUS^2
#define EPSv 1e-5f
#define INVR 10.0f        // 1 / RADIUS

typedef __bf16 v16bf __attribute__((ext_vector_type(16)));
typedef float  v8f   __attribute__((ext_vector_type(8)));

union FragBF { uint4 q[2]; v16bf v; };

__device__ __forceinline__ unsigned short f2bf(float f) {
    unsigned u = __float_as_uint(f);
    unsigned r = u + 0x7FFFu + ((u >> 16) & 1u);   // round-to-nearest-even
    return (unsigned short)(r >> 16);
}

// ---------------------------------------------------------------------------
// Prep: W (fp32, 256x131) -> bf16 padded [256 x 160]; fold bias+BN into
// per-channel scale/shift:  y = scale*acc + shift, scale = gamma*rsqrt(var+eps),
// shift = scale*(b - mean) + beta.
// ---------------------------------------------------------------------------
__global__ void k_prep(const float* __restrict__ W, const float* __restrict__ bias,
                       const float* __restrict__ gamma, const float* __restrict__ beta,
                       const float* __restrict__ mean, const float* __restrict__ var,
                       unsigned short* __restrict__ Wb,
                       float* __restrict__ scale, float* __restrict__ shift) {
    int g = blockIdx.x * 256 + threadIdx.x;
    if (g < DOUT * CINP) {
        int d = g / CINP, c = g % CINP;
        float v = (c < CIN) ? W[d * CIN + c] : 0.0f;
        Wb[g] = f2bf(v);
    }
    if (blockIdx.x == 0 && threadIdx.x < DOUT) {
        int d = threadIdx.x;
        float a = gamma[d] * rsqrtf(var[d] + EPSv);
        scale[d] = a;
        shift[d] = a * (bias[d] - mean[d]) + beta[d];
    }
}

// ---------------------------------------------------------------------------
// Farthest point sampling: 1 workgroup (1024 thr = 32 waves) per batch.
// Running min-dist in LDS; argmax with smallest-index tie-break via u64
// max-key = (dist_bits << 32) | (0xFFFFFFFF - idx).
// Also emits new_coor (B,3,S) directly into d_out.
// ---------------------------------------------------------------------------
__global__ __launch_bounds__(1024) void k_fps(const float* __restrict__ coor,
                                              int* __restrict__ fps,
                                              float* __restrict__ out) {
    __shared__ float dist[Nv];
    __shared__ unsigned long long wred[32];
    __shared__ unsigned long long win;
    int b = blockIdx.x, t = threadIdx.x;
    int lane = t & 31, wv = t >> 5;
    const float* xs = coor + (size_t)b * 3 * Nv;
    const float* ys = xs + Nv;
    const float* zs = ys + Nv;
    for (int p = t; p < Nv; p += 1024) dist[p] = 1e10f;
    __syncthreads();
    int far = 0;
    for (int i = 0; i < Sv; ++i) {
        if (t == 0) {
            fps[b * Sv + i] = far;
            out[(size_t)b * 3 * Sv + i]          = xs[far];
            out[(size_t)b * 3 * Sv + Sv + i]     = ys[far];
            out[(size_t)b * 3 * Sv + 2 * Sv + i] = zs[far];
        }
        float cx = xs[far], cy = ys[far], cz = zs[far];
        unsigned long long best = 0ull;
        for (int p = t; p < Nv; p += 1024) {
            float dx = xs[p] - cx, dy = ys[p] - cy, dz = zs[p] - cz;
            float dd = dx * dx + dy * dy + dz * dz;
            float nd = fminf(dist[p], dd);
            dist[p] = nd;
            unsigned long long key =
                ((unsigned long long)__float_as_uint(nd) << 32) |
                (unsigned long long)(0xFFFFFFFFu - (unsigned)p);
            best = key > best ? key : best;
        }
        #pragma unroll
        for (int m = 16; m >= 1; m >>= 1) {
            unsigned long long o = __shfl_xor(best, m);
            best = o > best ? o : best;
        }
        if (lane == 0) wred[wv] = best;
        __syncthreads();
        if (wv == 0) {
            unsigned long long k2 = wred[lane];
            #pragma unroll
            for (int m = 16; m >= 1; m >>= 1) {
                unsigned long long o = __shfl_xor(k2, m);
                k2 = o > k2 ? o : k2;
            }
            if (lane == 0) win = k2;
        }
        __syncthreads();
        far = (int)(0xFFFFFFFFu - (unsigned)win);
    }
}

// ---------------------------------------------------------------------------
// Hybrid knn / ball query: 1 workgroup (256 thr) per (b,s). All N distances in
// LDS; 32 rounds of exact argmin extraction (min-key ties on smallest index,
// matching top_k ordering); out-of-radius -> nearest.
// ---------------------------------------------------------------------------
__global__ __launch_bounds__(256) void k_knn(const float* __restrict__ coor,
                                             const int* __restrict__ fps,
                                             int* __restrict__ gidx) {
    __shared__ float dist[Nv];
    __shared__ unsigned long long wred[8];
    __shared__ int   sidx[Kv];
    __shared__ float sd[Kv];
    int bs = blockIdx.x;
    int b = bs >> 11;                  // / Sv
    int t = threadIdx.x, lane = t & 31, wv = t >> 5;
    const float* xs = coor + (size_t)b * 3 * Nv;
    const float* ys = xs + Nv;
    const float* zs = ys + Nv;
    int ci = fps[bs];
    float cx = xs[ci], cy = ys[ci], cz = zs[ci];
    for (int p = t; p < Nv; p += 256) {
        float dx = xs[p] - cx, dy = ys[p] - cy, dz = zs[p] - cz;
        dist[p] = dx * dx + dy * dy + dz * dz;
    }
    __syncthreads();
    for (int r = 0; r < Kv; ++r) {
        unsigned long long best = ~0ull;
        for (int p = t; p < Nv; p += 256) {
            unsigned long long key =
                ((unsigned long long)__float_as_uint(dist[p]) << 32) | (unsigned)p;
            best = key < best ? key : best;
        }
        #pragma unroll
        for (int m = 16; m >= 1; m >>= 1) {
            unsigned long long o = __shfl_xor(best, m);
            best = o < best ? o : best;
        }
        if (lane == 0) wred[wv] = best;
        __syncthreads();
        if (t == 0) {
            unsigned long long k2 = wred[0];
            #pragma unroll
            for (int w = 1; w < 8; ++w) k2 = wred[w] < k2 ? wred[w] : k2;
            int wi = (int)(unsigned)k2;
            sidx[r] = wi;
            sd[r]   = __uint_as_float((unsigned)(k2 >> 32));
            dist[wi] = __uint_as_float(0x7F800000u);  // +inf: remove
        }
        __syncthreads();
    }
    if (t < Kv) {
        int gi = (sd[t] > R2v) ? sidx[0] : sidx[t];
        gidx[(size_t)bs * Kv + t] = gi;
    }
}

__device__ __forceinline__ float chval(const float* feab, const float* xs, const float* ys,
                                       const float* zs, int n, float cx, float cy, float cz,
                                       int c) {
    if (c < Cv)     return feab[(size_t)c * Nv + n];
    if (c == Cv)     return (xs[n] - cx) * INVR;
    if (c == Cv + 1) return (ys[n] - cy) * INVR;
    if (c == Cv + 2) return (zs[n] - cz) * INVR;
    return 0.0f;
}

// ---------------------------------------------------------------------------
// Gather + bf16 WMMA GEMM (32x160 @ 160x256) + fused affine/ReLU/max-pool.
// 1 workgroup (256 thr = 8 waves) per (b,s); wave w owns N-tiles 2w, 2w+1.
// A (g matrix) staged in LDS as row-major bf16 pairs; B from pre-padded Wb.
// Fragment layouts per ISA 7.12.2 (16-bit A 16x32; B mirrors with lane=N col).
// ---------------------------------------------------------------------------
__global__ __launch_bounds__(256) void k_pn(const float* __restrict__ coor,
                                            const float* __restrict__ fea,
                                            const int* __restrict__ fps,
                                            const int* __restrict__ gidx,
                                            const unsigned short* __restrict__ Wb,
                                            const float* __restrict__ scale,
                                            const float* __restrict__ shift,
                                            float* __restrict__ out) {
    __shared__ unsigned int gA[Kv * 80];   // 32 rows x 160 bf16 (80 packed pairs)
    __shared__ int nidx[Kv];
    int bs = blockIdx.x;
    int b = bs >> 11;
    int s = bs & 2047;
    int t = threadIdx.x;
    if (t < Kv) nidx[t] = gidx[(size_t)bs * Kv + t];
    const float* xs = coor + (size_t)b * 3 * Nv;
    const float* ys = xs + Nv;
    const float* zs = ys + Nv;
    int ci = fps[bs];
    float cx = xs[ci], cy = ys[ci], cz = zs[ci];
    __syncthreads();

    const float* feab = fea + (size_t)b * Cv * Nv;
    for (int e = t; e < Kv * 80; e += 256) {
        int row = e / 80, pc = e % 80;
        int n = nidx[row];
        int c0 = pc * 2;
        float v0 = chval(feab, xs, ys, zs, n, cx, cy, cz, c0);
        float v1 = chval(feab, xs, ys, zs, n, cx, cy, cz, c0 + 1);
        gA[e] = (unsigned)f2bf(v0) | ((unsigned)f2bf(v1) << 16);
    }
    __syncthreads();

    int lane = t & 31, wv = t >> 5;
    int mlo = lane & 15, sel = lane >> 4;
    const uint4* wb4 = (const uint4*)Wb;          // 20 uint4 per output row
    float* ofea = out + (size_t)Bv * 3 * Sv;
    int arow0 = mlo * 80;
    int arow1 = (16 + mlo) * 80;

    for (int ntl = 0; ntl < 2; ++ntl) {
        int nt = wv * 2 + ntl;
        int ch = nt * 16 + mlo;
        v8f acc0 = {};
        v8f acc1 = {};
        int bbase = ch * 20 + sel;
        #pragma unroll
        for (int kc = 0; kc < 5; ++kc) {
            FragBF fa0, fa1, fb;
            int ao = kc * 16 + sel * 4;
            fa0.q[0] = *(const uint4*)&gA[arow0 + ao];
            fa0.q[1] = *(const uint4*)&gA[arow0 + ao + 8];
            fa1.q[0] = *(const uint4*)&gA[arow1 + ao];
            fa1.q[1] = *(const uint4*)&gA[arow1 + ao + 8];
            int bo = bbase + kc * 4;
            fb.q[0] = wb4[bo];
            fb.q[1] = wb4[bo + 2];
            if (kc < 4) __builtin_prefetch(&wb4[bo + 4], 0, 1);
            acc0 = __builtin_amdgcn_wmma_f32_16x16x32_bf16(
                false, fa0.v, false, fb.v, (short)0, acc0, false, false);
            acc1 = __builtin_amdgcn_wmma_f32_16x16x32_bf16(
                false, fa1.v, false, fb.v, (short)0, acc1, false, false);
        }
        // affine (scale may be negative -> apply before max), then max over
        // neighbors: 8 VGPR rows per half-lane, two M-tiles, lanes n / n+16.
        float sc = scale[ch], sh = shift[ch];
        float vmax = -3.0e38f;
        #pragma unroll
        for (int r = 0; r < 8; ++r) {
            vmax = fmaxf(vmax, fmaf(sc, acc0[r], sh));
            vmax = fmaxf(vmax, fmaf(sc, acc1[r], sh));
        }
        vmax = fmaxf(vmax, __shfl_xor(vmax, 16));
        vmax = fmaxf(vmax, 0.0f);                 // ReLU commutes with max
        if (sel == 0) ofea[((size_t)b * DOUT + ch) * Sv + s] = vmax;
    }
}

extern "C" void kernel_launch(void* const* d_in, const int* in_sizes, int n_in,
                              void* d_out, int out_size, void* d_ws, size_t ws_size,
                              hipStream_t stream) {
    const float* coor  = (const float*)d_in[0];
    const float* fea   = (const float*)d_in[1];
    const float* W     = (const float*)d_in[2];
    const float* bias  = (const float*)d_in[3];
    const float* gamma = (const float*)d_in[4];
    const float* beta  = (const float*)d_in[5];
    const float* mean  = (const float*)d_in[6];
    const float* var   = (const float*)d_in[7];
    float* out = (float*)d_out;

    char* ws = (char*)d_ws;
    int*            fpsb  = (int*)ws;                              // 4*2048 ints
    int*            gidx  = (int*)(ws + 32768);                    // 4*2048*32 ints
    unsigned short* Wb    = (unsigned short*)(ws + 32768 + 1048576);  // 256*160 bf16
    float*          scale = (float*)(ws + 32768 + 1048576 + 81920);
    float*          shift = scale + DOUT;

    hipLaunchKernelGGL(k_prep, dim3((DOUT * CINP + 255) / 256), dim3(256), 0, stream,
                       W, bias, gamma, beta, mean, var, Wb, scale, shift);
    hipLaunchKernelGGL(k_fps, dim3(Bv), dim3(1024), 0, stream, coor, fpsb, out);
    hipLaunchKernelGGL(k_knn, dim3(Bv * Sv), dim3(256), 0, stream, coor, fpsb, gidx);
    hipLaunchKernelGGL(k_pn, dim3(Bv * Sv), dim3(256), 0, stream,
                       coor, fea, fpsb, gidx, Wb, scale, shift, out);
}
